// SqueezeAndExcitationModel_90074054132601
// MI455X (gfx1250) — matrix operand order repaired and checked by reading, not compiled, not run on hardware
//
#include <hip/hip_runtime.h>
#include <hip/hip_bf16.h>

// ---------------------------------------------------------------------------
// MI455X (gfx1250, wave32) implementation.
// Convs are implicit GEMM on v_wmma_f32_16x16x32_f16 (f16 in, f32 accum).
// Activations: f16 NHWC, channels padded to multiples of 32.
// Weights: converted per launch to f16 [Cout_p][kh*kw][Cin_p].
// Conv kernel: WG tile = 64 Cout x 128 px, 8 waves, 4 WMMA / wave / K-step,
// global->reg prefetch pipelining across the LDS barrier.
// ---------------------------------------------------------------------------

typedef __attribute__((ext_vector_type(16))) _Float16 v16h;
typedef __attribute__((ext_vector_type(8)))  float    v8f;

union Frag16 { v16h v; uint4 q[2]; };
union H8     { uint4 q; _Float16 h[8]; };

struct ConvParams {
  const _Float16* __restrict__ in;     // NHWC f16, Cinp channels
  const _Float16* __restrict__ wg;     // [Coutp][KH*KW][Cinp] f16
  const float*    __restrict__ bias;   // [Cout] f32
  _Float16* __restrict__ out;          // NHWC f16, Coutp channels
  float*    __restrict__ outF32;       // if non-null: NCHW f32 (final layer)
  int N, H, W, Cinp, Cout, Coutp, KH, KW, pad, dil, relu;
};

// ---------------------------------------------------------------------------
// Implicit-GEMM conv. grid = (H, N, ceil(Coutp/64)), block = 256 (8 wave32).
// Wave (mt,nt) owns a 2x2 grid of 16x16 tiles: couts mt*32+{0,16},
// pixels nt*16+{0,64}. One image row (128 px) per workgroup.
// ---------------------------------------------------------------------------
__launch_bounds__(256)
__global__ void conv2d_wmma_f16(ConvParams P)
{
  __shared__ _Float16 As[64][40];    // [cout][k]   (+8 half row pad)
  __shared__ _Float16 Bs[128][40];   // [pixel][k]

  const int tid  = threadIdx.x;
  const int lane = tid & 31;
  const int wave = tid >> 5;
  const int mt = wave >> 2;            // cout block 0..1  (32 couts each)
  const int nt = wave & 3;             // pixel block 0..3 (16 px each half)
  const int h  = blockIdx.x;           // H = 128, full row per WG
  const int n  = blockIdx.y;
  const int coutBase = blockIdx.z * 64;

  const int l  = lane & 15;
  const int hs = lane >> 4;

  // staging coords: B 128px x 32k, 16 halves/thread; A 64rows x 32k, 8/thread
  const int bp = tid >> 1;             // B: pixel 0..127
  const int bc = (tid & 1) * 16;       // B: channel sub-offset (16 halves)
  const int arow = tid >> 2;           // A: cout row 0..63
  const int acol = (tid & 3) * 8;      // A: k sub-offset (8 halves)

  v8f acc[2][2] = {};

  const int Ktot  = P.KH * P.KW * P.Cinp;
  const int steps = Ktot >> 5;
  const size_t wrowBase = (size_t)(coutBase + arow) * (size_t)Ktot;
  const bool arowOk = (coutBase + arow) < P.Coutp;

  int c0 = 0, kh = 0, kw = 0;
  size_t tapOff = 0;                    // tap * Cinp

  uint4 rb0, rb1, ra0;
  // ---- prologue: load K-step 0 into registers ----
  {
    const int hi = h + kh * P.dil - P.pad;
    const int wi = bp + kw * P.dil - P.pad;
    rb0 = make_uint4(0u, 0u, 0u, 0u);
    rb1 = make_uint4(0u, 0u, 0u, 0u);
    if ((unsigned)hi < (unsigned)P.H && (unsigned)wi < (unsigned)P.W) {
      const _Float16* src = P.in + ((size_t)(n * P.H + hi) * P.W + wi) * P.Cinp + c0 + bc;
      rb0 = *(const uint4*)(src);
      rb1 = *(const uint4*)(src + 8);
    }
    ra0 = make_uint4(0u, 0u, 0u, 0u);
    if (arowOk) ra0 = *(const uint4*)(P.wg + wrowBase + tapOff + c0 + acol);
  }

  for (int s = 0; s < steps; ++s) {
    // ---- commit staged registers to LDS ----
    *(uint4*)(&Bs[bp][bc])     = rb0;
    *(uint4*)(&Bs[bp][bc + 8]) = rb1;
    *(uint4*)(&As[arow][acol]) = ra0;
    __syncthreads();

    // ---- advance (kh,kw,c0) and prefetch step s+1 into registers ----
    c0 += 32;
    if (c0 >= P.Cinp) {
      c0 = 0; tapOff += (size_t)P.Cinp;
      ++kw; if (kw == P.KW) { kw = 0; ++kh; }
    }
    if (s + 1 < steps) {
      const int hi = h + kh * P.dil - P.pad;
      const int wi = bp + kw * P.dil - P.pad;
      rb0 = make_uint4(0u, 0u, 0u, 0u);
      rb1 = make_uint4(0u, 0u, 0u, 0u);
      if ((unsigned)hi < (unsigned)P.H && (unsigned)wi < (unsigned)P.W) {
        const _Float16* src = P.in + ((size_t)(n * P.H + hi) * P.W + wi) * P.Cinp + c0 + bc;
        rb0 = *(const uint4*)(src);
        rb1 = *(const uint4*)(src + 8);
      }
      ra0 = make_uint4(0u, 0u, 0u, 0u);
      if (arowOk) ra0 = *(const uint4*)(P.wg + wrowBase + tapOff + c0 + acol);
    }

    // ---- fragments per ISA 7.12.2 wave32 layouts; 2x2 WMMAs ----
    Frag16 fa[2], fb[2];
    #pragma unroll
    for (int mi = 0; mi < 2; ++mi) {
      const _Float16* ap = &As[mt * 32 + mi * 16 + l][0];
      fa[mi].q[0] = *(const uint4*)(ap + hs * 8);        // K0..7   / K8..15
      fa[mi].q[1] = *(const uint4*)(ap + 16 + hs * 8);   // K16..23 / K24..31
    }
    #pragma unroll
    for (int ni = 0; ni < 2; ++ni) {
      const _Float16* bpp = &Bs[nt * 16 + ni * 64 + l][0];
      fb[ni].q[0] = *(const uint4*)(bpp + hs * 8);       // K0..7   / K8..15
      fb[ni].q[1] = *(const uint4*)(bpp + 16 + hs * 8);  // K16..23 / K24..31
    }
    #pragma unroll
    for (int mi = 0; mi < 2; ++mi)
      #pragma unroll
      for (int ni = 0; ni < 2; ++ni)
        acc[mi][ni] = __builtin_amdgcn_wmma_f32_16x16x32_f16(
            false, fa[mi].v, false, fb[ni].v, (short)0, acc[mi][ni], false, false);
    __syncthreads();
  }

  // ---- epilogue: bias (+ReLU) and store 2x2 tiles ----
  #pragma unroll
  for (int mi = 0; mi < 2; ++mi) {
    const int cw = coutBase + mt * 32 + mi * 16 + hs * 8;
    #pragma unroll
    for (int ni = 0; ni < 2; ++ni) {
      const int wpix = nt * 16 + ni * 64 + l;
      if (P.outF32) {
        #pragma unroll
        for (int r = 0; r < 8; ++r) {
          const int co = cw + r;
          if (co < P.Cout) {
            float v = acc[mi][ni][r] + P.bias[co];
            if (P.relu) v = fmaxf(v, 0.f);
            P.outF32[((size_t)(n * P.Cout + co) * P.H + h) * P.W + wpix] = v;
          }
        }
      } else if (cw < P.Coutp) {
        H8 o;
        #pragma unroll
        for (int r = 0; r < 8; ++r) {
          const int co = cw + r;
          float v = acc[mi][ni][r] + (co < P.Cout ? P.bias[co] : 0.f);
          if (P.relu) v = fmaxf(v, 0.f);
          o.h[r] = (_Float16)v;
        }
        *(uint4*)(P.out + ((size_t)(n * P.H + h) * P.W + wpix) * P.Coutp + cw) = o.q;
      }
    }
  }
}

// ---------------------------------------------------------------------------
// Normalize x, concat mask -> f16 NHWC [10][128][128][32] (ch 4..31 zero).
// ---------------------------------------------------------------------------
__launch_bounds__(256)
__global__ void prep_input_kernel(const float* __restrict__ x, const float* __restrict__ m,
                                  _Float16* __restrict__ act)
{
  const int idx = blockIdx.x * 256 + threadIdx.x;    // (n*16384 + hw)*32 + c
  if (idx >= 10 * 16384 * 32) return;
  const int c  = idx & 31;
  const int hw = (idx >> 5) & 16383;
  const int n  = idx >> 19;
  const int b = n / 5, f = n % 5;
  float v = 0.f;
  if (c < 3) {
    const float mean[3] = {0.485f, 0.456f, 0.406f};
    const float stdv[3] = {0.229f, 0.224f, 0.225f};
    v = (x[(((size_t)(b * 3 + c) * 5 + f) << 14) + hw] - mean[c]) / stdv[c];
  } else if (c == 3) {
    v = m[(((size_t)(b * 5 + f)) << 14) + hw];
  }
  act[idx] = (_Float16)v;
}

// ---------------------------------------------------------------------------
// OIHW f32 -> [Op][KHW][Ip] f16, zero-padded.
// ---------------------------------------------------------------------------
__launch_bounds__(256)
__global__ void prep_weights_kernel(const float* __restrict__ w, _Float16* __restrict__ wg,
                                    int O, int I, int KHW, int Op, int Ip)
{
  const int idx = blockIdx.x * 256 + threadIdx.x;
  if (idx >= Op * KHW * Ip) return;
  const int c = idx % Ip;
  const int t = (idx / Ip) % KHW;
  const int o = idx / (Ip * KHW);
  _Float16 v = (_Float16)0.f;
  if (o < O && c < I) v = (_Float16)w[((size_t)o * I + c) * KHW + t];
  wg[idx] = v;
}

// ---------------------------------------------------------------------------
// SE attention: per-pixel Conv3d(128,5)->64 + ReLU, Linear 64->5, softmax,
// writes att_map (fp32, into d_out) and dec_in (f16 NHWC [2][H][W][128]).
// ---------------------------------------------------------------------------
__launch_bounds__(256)
__global__ void attention_kernel(const _Float16* __restrict__ feats,  // [10][HW][128]
                                 const float* __restrict__ acw,       // (64,128,5)
                                 const float* __restrict__ acb,       // (64)
                                 const float* __restrict__ alw,       // (5,64)
                                 const float* __restrict__ alb,       // (5)
                                 float* __restrict__ attOut,          // (2,5,HW)
                                 _Float16* __restrict__ decIn)        // [2][HW][128]
{
  const int idx = blockIdx.x * 256 + threadIdx.x;
  if (idx >= 2 * 16384) return;
  const int b  = idx >> 14;
  const int hw = idx & 16383;

  float hid[64];
  #pragma unroll
  for (int o = 0; o < 64; ++o) hid[o] = acb[o];

  for (int f = 0; f < 5; ++f) {
    const _Float16* fp = feats + ((size_t)(b * 5 + f) * 16384 + hw) * 128;
    for (int c = 0; c < 128; c += 8) {
      H8 u; u.q = *(const uint4*)(fp + c);
      float v[8];
      #pragma unroll
      for (int j = 0; j < 8; ++j) v[j] = (float)u.h[j];
      const float* wr = acw + (size_t)c * 5 + f;
      #pragma unroll
      for (int o = 0; o < 64; ++o) {
        const float* w8 = wr + (size_t)o * 640;
        float s = 0.f;
        #pragma unroll
        for (int j = 0; j < 8; ++j) s += v[j] * w8[j * 5];
        hid[o] += s;
      }
    }
  }
  #pragma unroll
  for (int o = 0; o < 64; ++o) hid[o] = fmaxf(hid[o], 0.f);

  float lg[5];
  #pragma unroll
  for (int j = 0; j < 5; ++j) {
    float s = alb[j];
    #pragma unroll
    for (int o = 0; o < 64; ++o) s += alw[j * 64 + o] * hid[o];
    lg[j] = s;
  }
  float mx = lg[0];
  #pragma unroll
  for (int j = 1; j < 5; ++j) mx = fmaxf(mx, lg[j]);
  float e[5], den = 0.f;
  #pragma unroll
  for (int j = 0; j < 5; ++j) { e[j] = expf(lg[j] - mx); den += e[j]; }
  const float inv = 1.f / den;
  float att[5];
  #pragma unroll
  for (int j = 0; j < 5; ++j) {
    att[j] = e[j] * inv;
    attOut[(((size_t)(b * 5 + j)) << 14) + hw] = att[j];
  }

  _Float16* dp = decIn + (size_t)idx * 128;
  for (int c = 0; c < 128; c += 8) {
    float a8[8] = {0.f, 0.f, 0.f, 0.f, 0.f, 0.f, 0.f, 0.f};
    #pragma unroll
    for (int f = 0; f < 5; ++f) {
      const _Float16* fp = feats + ((size_t)(b * 5 + f) * 16384 + hw) * 128;
      H8 u; u.q = *(const uint4*)(fp + c);
      #pragma unroll
      for (int j = 0; j < 8; ++j) a8[j] += att[f] * (float)u.h[j];
    }
    H8 o;
    #pragma unroll
    for (int j = 0; j < 8; ++j) o.h[j] = (_Float16)a8[j];
    *(uint4*)(dp + c) = o.q;
  }
}

// ---------------------------------------------------------------------------
static inline int rup(int x, int a) { return (x + a - 1) / a * a; }

extern "C" void kernel_launch(void* const* d_in, const int* in_sizes, int n_in,
                              void* d_out, int out_size, void* d_ws, size_t ws_size,
                              hipStream_t stream)
{
  (void)in_sizes; (void)n_in; (void)out_size; (void)ws_size;

  // d_in order: 0:x 1:m 2:y 3:t 4:r_list 5-9:enc_ws 10-14:enc_bs
  //             15:att_conv_w 16:att_conv_b 17:att_lin_w 18:att_lin_b
  //             19-31:dec_ws 32-44:dec_bs
  const float* x = (const float*)d_in[0];
  const float* m = (const float*)d_in[1];
  const float* att_conv_w = (const float*)d_in[15];
  const float* att_conv_b = (const float*)d_in[16];
  const float* att_lin_w  = (const float*)d_in[17];
  const float* att_lin_b  = (const float*)d_in[18];

  struct LD { int wi, bi, Cin, Cout, K, pad, dil, relu; };
  const LD L[18] = {
    // encoder (N=10)
    {5,10,  4, 64, 5, 2, 1, 1}, {6,11, 64, 64, 3, 1, 1, 1},
    {7,12, 64,128, 3, 1, 1, 1}, {8,13,128,128, 3, 1, 1, 1},
    {9,14,128,128, 3, 1, 1, 0},
    // decoder (N=2)
    {19,32,128,257,3, 1, 1,1}, {20,33,257,257,3, 1, 1,1}, {21,34,257,257,3, 1, 1,1},
    {22,35,257,257,3, 2, 2,1}, {23,36,257,257,3, 4, 4,1}, {24,37,257,257,3, 8, 8,1},
    {25,38,257,257,3,16,16,1}, {26,39,257,257,3, 1, 1,1}, {27,40,257,128,3, 1, 1,1},
    {28,41,128,128,3, 1, 1,1}, {29,42,128, 64,3, 1, 1,1}, {30,43, 64, 64,3, 1, 1,1},
    {31,44, 64,  3,5, 2, 1,0}};

  char* ws = (char*)d_ws;
  size_t off = 0;
  auto carve = [&](size_t bytes) -> void* {
    void* p = ws + off;
    off += (bytes + 255) & ~(size_t)255;
    return p;
  };

  const size_t actElems = (size_t)10 * 16384 * 128;      // max NHWC footprint
  _Float16* bufs[2];
  bufs[0] = (_Float16*)carve(actElems * 2);
  bufs[1] = (_Float16*)carve(actElems * 2);

  _Float16* wg[18];
  int Cinp[18], Coutp[18];
  for (int i = 0; i < 18; ++i) {
    Cinp[i]  = rup(L[i].Cin, 32);
    Coutp[i] = (i == 17) ? 16 : rup(L[i].Cout, 32);
    wg[i] = (_Float16*)carve((size_t)Coutp[i] * L[i].K * L[i].K * Cinp[i] * 2);
  }

  // --- weight conversion ---
  for (int i = 0; i < 18; ++i) {
    const int KHW = L[i].K * L[i].K;
    const int total = Coutp[i] * KHW * Cinp[i];
    prep_weights_kernel<<<(total + 255) / 256, 256, 0, stream>>>(
        (const float*)d_in[L[i].wi], wg[i], L[i].Cout, L[i].Cin, KHW, Coutp[i], Cinp[i]);
  }

  // --- encoder input prep ---
  {
    const int total = 10 * 16384 * 32;
    prep_input_kernel<<<(total + 255) / 256, 256, 0, stream>>>(x, m, bufs[0]);
  }

  float* y_hat  = (float*)d_out;                 // (2,3,128,128) NCHW
  float* attOut = (float*)d_out + 2 * 3 * 16384; // (2,1,5,128,128)

  auto runConv = [&](int i, const _Float16* src, _Float16* dst, float* dstF32, int N) {
    ConvParams P;
    P.in = src; P.wg = wg[i]; P.bias = (const float*)d_in[L[i].bi];
    P.out = dst; P.outF32 = dstF32;
    P.N = N; P.H = 128; P.W = 128;
    P.Cinp = Cinp[i]; P.Cout = L[i].Cout; P.Coutp = Coutp[i];
    P.KH = L[i].K; P.KW = L[i].K; P.pad = L[i].pad; P.dil = L[i].dil; P.relu = L[i].relu;
    dim3 grid(128 /* one row per WG */, N, (Coutp[i] + 63) / 64);
    conv2d_wmma_f16<<<grid, 256, 0, stream>>>(P);
  };

  // --- encoder: ping-pong, c_feats ends in bufs[1] ---
  runConv(0, bufs[0], bufs[1], nullptr, 10);
  runConv(1, bufs[1], bufs[0], nullptr, 10);
  runConv(2, bufs[0], bufs[1], nullptr, 10);
  runConv(3, bufs[1], bufs[0], nullptr, 10);
  runConv(4, bufs[0], bufs[1], nullptr, 10);

  // --- attention: reads c_feats (bufs[1]), writes dec_in -> bufs[0] ---
  attention_kernel<<<(2 * 16384 + 255) / 256, 256, 0, stream>>>(
      bufs[1], att_conv_w, att_conv_b, att_lin_w, att_lin_b, attOut, bufs[0]);

  // --- decoder: 12 f16 layers ping-pong, final layer -> fp32 NCHW d_out ---
  int cur = 0;
  for (int i = 5; i < 17; ++i) {
    runConv(i, bufs[cur], bufs[1 - cur], nullptr, 2);
    cur = 1 - cur;
  }
  runConv(17, bufs[cur], nullptr, y_hat, 2);
}
